// GatedNorm_56023553409270
// MI455X (gfx1250) — matrix operand order repaired and checked by reading, not compiled, not run on hardware
//
#include <hip/hip_runtime.h>

typedef __attribute__((ext_vector_type(16))) __bf16 v16bf;
typedef __attribute__((ext_vector_type(8)))  __bf16 v8bf;
typedef __attribute__((ext_vector_type(8)))  float  v8f;

#define DIM 4096
#define RNK 128
#define NTOK 16384
#define WAVES 8
#define TOK_PER_WAVE 16
#define TOK_PER_WG (WAVES * TOK_PER_WAVE)   // 128
#define KSLAB 128
#define EPS 1.1920929e-07f

static __device__ __forceinline__ v16bf join8(v8bf a, v8bf b) {
  return __builtin_shufflevector(a, b, 0,1,2,3,4,5,6,7,8,9,10,11,12,13,14,15);
}

__global__ __launch_bounds__(256) void gated_norm_kernel(
    const float* __restrict__ x, const float* __restrict__ Wd,
    const float* __restrict__ Wu, float* __restrict__ out)
{
  // 32 KB weight slab (bf16, reused between GEMM1/GEMM2), 32 KB h, scales
  __shared__ __align__(16) __bf16 sW[128 * 128];
  __shared__ __align__(16) __bf16 sH[WAVES * 16 * 128];
  __shared__ float sScale[TOK_PER_WG];

  const int tid  = threadIdx.x;
  const int lane = tid & 31;
  const int w    = tid >> 5;
  const int lo   = lane & 15;
  const int hi   = lane >> 4;
  const int tokBase = blockIdx.x * TOK_PER_WG + w * TOK_PER_WAVE;

  // ---- Phase 1: RMS scale per token (coalesced, wave-cooperative) ----
  for (int t = 0; t < TOK_PER_WAVE; ++t) {
    const float4* row = (const float4*)(x + (size_t)(tokBase + t) * DIM);
    float ss = 0.f;
    #pragma unroll 4
    for (int i = lane; i < DIM / 4; i += 32) {
      float4 v = row[i];
      ss += v.x * v.x + v.y * v.y + v.z * v.z + v.w * v.w;
    }
    #pragma unroll
    for (int m = 16; m >= 1; m >>= 1) ss += __shfl_xor(ss, m, 32);
    if (lane == 0) sScale[w * 16 + t] = rsqrtf(ss * (1.0f / DIM) + EPS);
  }
  const float sA = sScale[w * 16 + lo];  // scale for this lane's A-row token

  // ---- GEMM1: h = silu(xn @ Wd^T), 8 accumulator tiles cover R=128 ----
  v8f acc[8];
  #pragma unroll
  for (int i = 0; i < 8; ++i) acc[i] = v8f{0.f,0.f,0.f,0.f,0.f,0.f,0.f,0.f};

  const float* xrow = x + (size_t)(tokBase + lo) * DIM;

  for (int ks = 0; ks < DIM; ks += KSLAB) {
    __syncthreads();
    // stage Wd[:, ks:ks+128) as bf16 (all 256 threads)
    for (int idx = tid; idx < 128 * 32; idx += 256) {
      int r = idx >> 5;
      int c = (idx & 31) << 2;
      float4 f = *(const float4*)(Wd + (size_t)r * DIM + ks + c);
      __bf16* p = &sW[r * 128 + c];
      p[0] = (__bf16)f.x; p[1] = (__bf16)f.y; p[2] = (__bf16)f.z; p[3] = (__bf16)f.w;
    }
    __syncthreads();

    #pragma unroll
    for (int kc = 0; kc < KSLAB / 32; ++kc) {
      const int kb = ks + kc * 32;
      // A fragment: 16x32 bf16 per ISA layout (two 8-elem K runs per lane)
      const int k0 = kb + hi * 8;
      const int k1 = kb + 16 + hi * 8;
      float4 f0 = *(const float4*)(xrow + k0);
      float4 f1 = *(const float4*)(xrow + k0 + 4);
      float4 f2 = *(const float4*)(xrow + k1);
      float4 f3 = *(const float4*)(xrow + k1 + 4);
      v16bf a;
      a[0]=(__bf16)(f0.x*sA);  a[1]=(__bf16)(f0.y*sA);  a[2]=(__bf16)(f0.z*sA);  a[3]=(__bf16)(f0.w*sA);
      a[4]=(__bf16)(f1.x*sA);  a[5]=(__bf16)(f1.y*sA);  a[6]=(__bf16)(f1.z*sA);  a[7]=(__bf16)(f1.w*sA);
      a[8]=(__bf16)(f2.x*sA);  a[9]=(__bf16)(f2.y*sA);  a[10]=(__bf16)(f2.z*sA); a[11]=(__bf16)(f2.w*sA);
      a[12]=(__bf16)(f3.x*sA); a[13]=(__bf16)(f3.y*sA); a[14]=(__bf16)(f3.z*sA); a[15]=(__bf16)(f3.w*sA);

      const int kloc = kc * 32 + hi * 8;
      #pragma unroll
      for (int nt = 0; nt < 8; ++nt) {
        const __bf16* bp = &sW[(nt * 16 + lo) * 128 + kloc];
        v16bf b = join8(*(const v8bf*)bp, *(const v8bf*)(bp + 16));
        acc[nt] = __builtin_amdgcn_wmma_f32_16x16x32_bf16(
            false, a, false, b, (short)0, acc[nt], false, false);
      }
    }
  }

  // ---- SiLU, write h (bf16) to per-wave LDS region ----
  {
    __bf16* hb = &sH[w * 16 * 128];
    #pragma unroll
    for (int nt = 0; nt < 8; ++nt) {
      #pragma unroll
      for (int v = 0; v < 8; ++v) {
        float hv = acc[nt][v];
        float s = hv / (1.0f + __expf(-hv));
        hb[(v + hi * 8) * 128 + nt * 16 + lo] = (__bf16)s;  // row=token, col=r
      }
    }
  }

  // ---- A fragments for GEMM2 (K = 128 -> 4 chunks), same-wave LDS reload ----
  v16bf ah[4];
  {
    const __bf16* hrow = &sH[w * 16 * 128 + lo * 128];
    #pragma unroll
    for (int kc = 0; kc < 4; ++kc) {
      const int k0 = kc * 32 + hi * 8;
      ah[kc] = join8(*(const v8bf*)(hrow + k0), *(const v8bf*)(hrow + k0 + 16));
    }
  }

  // per-lane output scales for tokens v + hi*8 (C/D layout)
  float scv[8];
  #pragma unroll
  for (int v = 0; v < 8; ++v) scv[v] = sScale[w * 16 + v + hi * 8];

  // ---- GEMM2: gate = sigmoid(h @ Wu^T); out = xn * gate ----
  for (int ns = 0; ns < DIM; ns += 128) {
    __syncthreads();
    // stage Wu[ns:ns+128)[0:128) as bf16
    for (int idx = tid; idx < 128 * 32; idx += 256) {
      int r = idx >> 5;
      int c = (idx & 31) << 2;
      float4 f = *(const float4*)(Wu + (size_t)(ns + r) * RNK + c);
      __bf16* p = &sW[r * 128 + c];
      p[0] = (__bf16)f.x; p[1] = (__bf16)f.y; p[2] = (__bf16)f.z; p[3] = (__bf16)f.w;
    }
    __syncthreads();

    #pragma unroll
    for (int ntl = 0; ntl < 8; ++ntl) {
      const int n0 = ns + ntl * 16;
      v8f g = v8f{0.f,0.f,0.f,0.f,0.f,0.f,0.f,0.f};
      #pragma unroll
      for (int kc = 0; kc < 4; ++kc) {
        const __bf16* bp = &sW[(ntl * 16 + lo) * 128 + kc * 32 + hi * 8];
        v16bf b = join8(*(const v8bf*)bp, *(const v8bf*)(bp + 16));
        g = __builtin_amdgcn_wmma_f32_16x16x32_bf16(
            false, ah[kc], false, b, (short)0, g, false, false);
      }
      const int d = n0 + lo;
      #pragma unroll
      for (int v = 0; v < 8; ++v) {
        float gate = 1.0f / (1.0f + __expf(-g[v]));
        size_t off = (size_t)(tokBase + v + hi * 8) * DIM + d;
        out[off] = x[off] * scv[v] * gate;   // xn = x * scale (fp32 precision)
      }
    }
  }
}

extern "C" void kernel_launch(void* const* d_in, const int* in_sizes, int n_in,
                              void* d_out, int out_size, void* d_ws, size_t ws_size,
                              hipStream_t stream) {
  const float* x  = (const float*)d_in[0];
  const float* Wd = (const float*)d_in[1];
  const float* Wu = (const float*)d_in[2];
  float* out = (float*)d_out;
  (void)in_sizes; (void)n_in; (void)out_size; (void)d_ws; (void)ws_size;
  gated_norm_kernel<<<NTOK / TOK_PER_WG, 256, 0, stream>>>(x, Wd, Wu, out);
}
